// LSTMBlock_88502096101699
// MI455X (gfx1250) — matrix operand order repaired and checked by reading, not compiled, not run on hardware
//
#include <hip/hip_runtime.h>
#include <hip/hip_bf16.h>

// Problem constants
#define kB 64
#define kI 128
#define kT 1024
#define kH 256
#define kG 1024   // 4*kH
#define NWG 16    // workgroups in persistent recurrent kernel; each owns kH/NWG=16 h cols
#define KTOT 384  // kI + kH fused K dimension

typedef __attribute__((ext_vector_type(16))) __bf16 v16bf;
typedef __attribute__((ext_vector_type(8)))  float  v8f;

union ABfrag {
    v16bf v;
    uint4 u[2];
};

// ---------------------------------------------------------------------------
// init: bias = b_ih + b_hh, zero both h ping-pong buffers, reset barrier cnt
// grid: 64 x 256
// ---------------------------------------------------------------------------
__global__ void init_ws_kernel(const float* __restrict__ b_ih,
                               const float* __restrict__ b_hh,
                               float* __restrict__ bias,
                               __bf16* __restrict__ hbuf,   // [2][kB][kH]
                               unsigned* __restrict__ cnt) {
    int gid = blockIdx.x * 256 + threadIdx.x;           // 0..16383
    if (gid < kG) bias[gid] = b_ih[gid] + b_hh[gid];
    // zero 2*64*256 = 32768 bf16 -> 2 per thread
    hbuf[gid * 2 + 0] = (__bf16)0.0f;
    hbuf[gid * 2 + 1] = (__bf16)0.0f;
    if (gid == 0) *cnt = 0u;
}

// ---------------------------------------------------------------------------
// pack x: [B][I][T] f32  ->  xT: [T][B][I] bf16 (per-t 64x128 row-major tile)
// grid: (T/32, I/32, B) x 256, LDS tile transpose for coalescing both sides
// ---------------------------------------------------------------------------
__global__ void pack_x_kernel(const float* __restrict__ x,
                              __bf16* __restrict__ xT) {
    __shared__ float tile[32][33];
    const int t0 = blockIdx.x * 32;
    const int i0 = blockIdx.y * 32;
    const int b  = blockIdx.z;
    for (int idx = threadIdx.x; idx < 1024; idx += 256) {
        int il = idx >> 5, tl = idx & 31;                // consecutive tid -> consecutive t
        tile[il][tl] = x[((size_t)b * kI + (i0 + il)) * kT + (t0 + tl)];
    }
    __syncthreads();
    for (int idx = threadIdx.x; idx < 1024; idx += 256) {
        int tl = idx >> 5, il = idx & 31;                // consecutive tid -> consecutive i
        xT[((size_t)(t0 + tl) * kB + b) * kI + (i0 + il)] = (__bf16)tile[il][tl];
    }
}

// ---------------------------------------------------------------------------
// Persistent fused LSTM recurrence.
//   gates[64 x 64cols] = [x_t | h_prev](64x384) x Wl^T(384x64) via bf16 WMMA
//   Wl (48KB bf16) resident in LDS for all T steps. c-state in LDS.
//   Grid-wide step barrier via atomic counter.
// grid: NWG x 256 (8 waves; wave w -> m-tile w&3, n-tiles (w>>2)*2, +1)
// dynamic LDS: Wl 49152 | gbuf 64*68*4 | cbuf 64*16*4 | biasl 64*4
// ---------------------------------------------------------------------------
__global__ __launch_bounds__(256) void lstm_persistent_kernel(
    const float* __restrict__ W_ih,     // [kG][kI]
    const float* __restrict__ W_hh,     // [kG][kH]
    const __bf16* __restrict__ xT,      // [kT][kB][kI]
    const float* __restrict__ bias,     // [kG]
    __bf16* __restrict__ hbuf,          // [2][kB][kH]
    unsigned* __restrict__ cnt,
    float* __restrict__ out)            // [kB][kH][kT]
{
    extern __shared__ char smem[];
    __bf16 (*Wl)[KTOT] = (__bf16(*)[KTOT])smem;              // 64 x 384 bf16 = 49152B
    float* gbuf  = (float*)(smem + 49152);                   // 64 x 68 f32   = 17408B
    float* cbuf  = (float*)(smem + 49152 + 17408);           // 64 x 16 f32   =  4096B
    float* biasl = (float*)(smem + 49152 + 17408 + 4096);    // 64 f32

    const int tid = threadIdx.x;
    const int jo  = blockIdx.x * 16;   // this WG owns h columns [jo, jo+16)

    // Load + convert this WG's weight slice: row n = q*16+j  <->  gate row q*kH+jo+j
    for (int idx = tid; idx < 64 * KTOT; idx += 256) {
        int n = idx / KTOT, k = idx - n * KTOT;
        int g = (n >> 4) * kH + jo + (n & 15);
        float w = (k < kI) ? W_ih[(size_t)g * kI + k]
                           : W_hh[(size_t)g * kH + (k - kI)];
        Wl[n][k] = (__bf16)w;
    }
    if (tid < 64) biasl[tid] = bias[(tid >> 4) * kH + jo + (tid & 15)];
    for (int idx = tid; idx < 64 * 16; idx += 256) cbuf[idx] = 0.0f;
    __syncthreads();

    const int lane  = tid & 31;
    const int wave  = tid >> 5;
    const int mt    = wave & 3;          // m tile: batch rows [mt*16, mt*16+16)
    const int nt0   = (wave >> 2) * 2;   // n tiles nt0, nt0+1 (of 4 = gates i,f,g,o)
    const int nl    = lane & 15;
    const int khalf = lane >> 4;

    for (int t = 0; t < kT; ++t) {
        const __bf16* xa = xT   + (size_t)t * (kB * kI);            // A cols 0..127
        const __bf16* ha = hbuf + (size_t)(t & 1) * (kB * kH);      // A cols 128..383
        __bf16*       hn = hbuf + (size_t)((t + 1) & 1) * (kB * kH);

        v8f acc0 = {};
        v8f acc1 = {};
        const int arow = mt * 16 + nl;   // A-matrix row (batch index) for this lane

        #pragma unroll
        for (int kk = 0; kk < KTOT / 32; ++kk) {
            const int kb = kk * 32;
            // A fragment (16x32 bf16): 2 x b128 per lane, documented layout
            ABfrag a;
            const __bf16* ap = (kb < kI)
                ? (xa + (size_t)arow * kI + kb        + 8 * khalf)
                : (ha + (size_t)arow * kH + (kb - kI) + 8 * khalf);
            a.u[0] = *(const uint4*)(ap);        // VGPR0-3: K = kb + 8*khalf + 0..7
            a.u[1] = *(const uint4*)(ap + 16);   // VGPR4-7: K = kb + 16 + 8*khalf + 0..7
            // B fragments (32x16 bf16) from LDS weight slice: 2 x ds b128 per lane
            ABfrag b0, b1;
            const __bf16* bp0 = &Wl[nt0 * 16 + nl][kb + 8 * khalf];
            const __bf16* bp1 = &Wl[(nt0 + 1) * 16 + nl][kb + 8 * khalf];
            b0.u[0] = *(const uint4*)(bp0);
            b0.u[1] = *(const uint4*)(bp0 + 16);
            b1.u[0] = *(const uint4*)(bp1);
            b1.u[1] = *(const uint4*)(bp1 + 16);

            acc0 = __builtin_amdgcn_wmma_f32_16x16x32_bf16(false, a.v, false, b0.v,
                                                           (short)0, acc0, false, false);
            acc1 = __builtin_amdgcn_wmma_f32_16x16x32_bf16(false, a.v, false, b1.v,
                                                           (short)0, acc1, false, false);
        }

        // Prefetch next step's x tile (16KB = 128 cachelines of 128B)
        if (t + 1 < kT && tid < 128) {
            __builtin_prefetch((const void*)(xT + (size_t)(t + 1) * (kB * kI) + tid * 64), 0, 1);
        }

        // Scatter D (f32 16x16 layout) to padded LDS gate buffer
        #pragma unroll
        for (int r = 0; r < 8; ++r) {
            int mrow = mt * 16 + khalf * 8 + r;
            gbuf[mrow * 68 + nt0 * 16 + nl]       = acc0[r];
            gbuf[mrow * 68 + (nt0 + 1) * 16 + nl] = acc1[r];
        }
        __syncthreads();

        // Fused cell update: 64 batches x 16 cols, 4 items/thread
        #pragma unroll
        for (int r = 0; r < 4; ++r) {
            int item = tid + r * 256;        // = b*16 + j
            int bb = item >> 4, j = item & 15;
            float iv = gbuf[bb * 68 +      j] + biasl[j];
            float fv = gbuf[bb * 68 + 16 + j] + biasl[16 + j];
            float gv = gbuf[bb * 68 + 32 + j] + biasl[32 + j];
            float ov = gbuf[bb * 68 + 48 + j] + biasl[48 + j];
            iv = 1.0f / (1.0f + __expf(-iv));
            fv = 1.0f / (1.0f + __expf(-fv));
            gv = tanhf(gv);
            ov = 1.0f / (1.0f + __expf(-ov));
            float c = fv * cbuf[item] + iv * gv;
            cbuf[item] = c;
            float h = ov * tanhf(c);
            hn[bb * kH + jo + j] = (__bf16)h;                       // next-step h (bf16)
            out[((size_t)bb * kH + (jo + j)) * kT + t] = fmaxf(h, 0.0f);  // relu output
        }

        // Grid-wide step barrier (monotonic counter, max 16*1024 = 16384)
        __threadfence();
        __syncthreads();
        if (tid == 0) {
            atomicAdd(cnt, 1u);
            const unsigned target = (unsigned)NWG * (unsigned)(t + 1);
            while (__hip_atomic_load(cnt, __ATOMIC_ACQUIRE, __HIP_MEMORY_SCOPE_AGENT) < target) {
                __builtin_amdgcn_s_sleep(8);
            }
        }
        __syncthreads();
    }
}

// ---------------------------------------------------------------------------
// Workspace layout (bytes):
//   [0]        counter (4B, padded to 256)
//   [256]      bias f32 [1024]                  (4KB)
//   [8192]     hbuf bf16 [2][64][256]           (64KB)
//   [131072]   xT bf16 [1024][64][128]          (16MB)
// total ~16.2MB
// ---------------------------------------------------------------------------
extern "C" void kernel_launch(void* const* d_in, const int* in_sizes, int n_in,
                              void* d_out, int out_size, void* d_ws, size_t ws_size,
                              hipStream_t stream) {
    const float* x    = (const float*)d_in[0];   // [64][128][1024]
    const float* W_ih = (const float*)d_in[1];   // [1024][128]
    const float* W_hh = (const float*)d_in[2];   // [1024][256]
    const float* b_ih = (const float*)d_in[3];   // [1024]
    const float* b_hh = (const float*)d_in[4];   // [1024]
    float* out = (float*)d_out;                  // [64][256][1024]

    char* ws = (char*)d_ws;
    unsigned* cnt = (unsigned*)ws;
    float*    bias = (float*)(ws + 256);
    __bf16*   hbuf = (__bf16*)(ws + 8192);
    __bf16*   xT   = (__bf16*)(ws + 131072);

    init_ws_kernel<<<64, 256, 0, stream>>>(b_ih, b_hh, bias, hbuf, cnt);

    dim3 pgrid(kT / 32, kI / 32, kB);
    pack_x_kernel<<<pgrid, 256, 0, stream>>>(x, xT);

    const size_t lds_bytes = 49152 + 17408 + 4096 + 256;  // ~71KB dynamic LDS
    lstm_persistent_kernel<<<NWG, 256, lds_bytes, stream>>>(W_ih, W_hh, xT, bias,
                                                            hbuf, cnt, out);
}
